// GT_20246475833913
// MI455X (gfx1250) — compile-verified, matrix operand order checked by reading, not run on hardware
//
#include <hip/hip_runtime.h>

// ---------- types ----------
typedef __bf16 bf16_t;
typedef __attribute__((ext_vector_type(16))) __bf16 v16bf;
typedef __attribute__((ext_vector_type(8)))  __bf16 v8bf;
typedef __attribute__((ext_vector_type(8)))  float  v8f;
typedef __attribute__((ext_vector_type(4)))  float  v4f;

#define NTOK 1024
#define HID  512
#define NH   16
#define DH   32
#define KED  4
#define ED   32

__device__ __forceinline__ v8f wmma_bf16(v16bf a, v16bf b, v8f c) {
  return __builtin_amdgcn_wmma_f32_16x16x32_bf16(false, a, false, b, (short)0, c, false, false);
}

// Load a 16-lane fragment: per-lane two contiguous 8-half (16B) chunks at
// kk + (hi?8:0) and kk + 16 + (hi?8:0).  Matches CDNA5 16-bit A/B layout.
__device__ __forceinline__ v16bf load_frag_b(const bf16_t* __restrict__ rowptr, int kk, int hi) {
  const bf16_t* p = rowptr + kk + (hi ? 8 : 0);
  v8bf a = *(const v8bf*)(p);
  v8bf b = *(const v8bf*)(p + 16);
  v16bf r;
#pragma unroll
  for (int i = 0; i < 8; ++i) { r[i] = a[i]; r[i + 8] = b[i]; }
  return r;
}

// ---------- 1. centrality encoding: h = x + in_emb[deg] + out_emb[deg] ----------
__global__ __launch_bounds__(128) void k_centrality(
    const float* __restrict__ x, const int* __restrict__ ind, const int* __restrict__ outd,
    const float* __restrict__ inde, const float* __restrict__ outde,
    float* __restrict__ h, bf16_t* __restrict__ hbf) {
  int n = blockIdx.x, t = threadIdx.x;
  int di = ind[n], dn = outd[n];
#pragma unroll
  for (int j = 0; j < 4; ++j) {
    int c = t + j * 128;
    float v = x[n * HID + c] + inde[di * HID + c] + outde[dn * HID + c];
    h[n * HID + c] = v;
    hbf[n * HID + c] = (bf16_t)v;
  }
}

// ---------- 2. weight transpose + bf16 convert: Wt[o][i] = W[i][o] ----------
__global__ __launch_bounds__(256) void k_wt(const float* __restrict__ W, bf16_t* __restrict__ Wt,
                                            int Kin, int Nout) {
  int idx = blockIdx.x * 256 + threadIdx.x;
  if (idx >= Kin * Nout) return;
  int o = idx / Kin, i = idx % Kin;
  Wt[idx] = (bf16_t)W[i * Nout + o];
}

// pos_w transpose: pwT[h][k*32+d] = edge_pos_emb[k][d*16+h]
__global__ __launch_bounds__(256) void k_pw(const float* __restrict__ epw, bf16_t* __restrict__ pwT) {
  int idx = blockIdx.x * 256 + threadIdx.x;
  if (idx >= NH * 128) return;
  int hh = idx >> 7, kd = idx & 127, kq = kd >> 5, d = kd & 31;
  pwT[idx] = (bf16_t)epw[kq * (ED * NH) + d * NH + hh];
}

// ---------- 3. attention bias via WMMA: bias[h][i][j] ----------
__global__ __launch_bounds__(256) void k_bias(
    const int* __restrict__ spos, const int* __restrict__ eidx,
    const float* __restrict__ semb, const float* __restrict__ eemb,
    const bf16_t* __restrict__ pwT, float* __restrict__ bias) {
  int lane = threadIdx.x & 31, wave = threadIdx.x >> 5;
  int task = blockIdx.x * 8 + wave;            // 1024 * 64 tasks
  int i = task >> 6, j0 = (task & 63) << 4;
  int hi = lane >> 4, l4 = lane & 15;
  int jj = j0 + l4;                            // this lane's pair row
  const bf16_t* brow = pwT + l4 * 128;         // B row = head l4
  v8f acc0 = {}, acc1 = {};                    // split D->C chains
  int cnt = 0;
#pragma unroll
  for (int s = 0; s < KED; ++s) {
    int idx = eidx[((long)i * NTOK + jj) * KED + s];
    cnt += (idx != 0);
    const float* ep = eemb + (long)idx * ED + (hi ? 8 : 0);
    v4f f0 = *(const v4f*)(ep);
    v4f f1 = *(const v4f*)(ep + 4);
    v4f f2 = *(const v4f*)(ep + 16);
    v4f f3 = *(const v4f*)(ep + 20);
    v16bf a;
#pragma unroll
    for (int q = 0; q < 4; ++q) {
      a[q] = (bf16_t)f0[q];  a[q + 4]  = (bf16_t)f1[q];
      a[q + 8] = (bf16_t)f2[q]; a[q + 12] = (bf16_t)f3[q];
    }
    v16bf b = load_frag_b(brow, s * 32, hi);
    if (s & 1) acc1 = wmma_bf16(a, b, acc1);
    else       acc0 = wmma_bf16(a, b, acc0);
  }
  float fcnt = (float)cnt;
#pragma unroll
  for (int r = 0; r < 8; ++r) {
    int p = r + (hi ? 8 : 0);
    float pl = __shfl(fcnt, hi ? (8 + r) : r, 16);
    pl = pl < 1.f ? 1.f : pl;
    int j = j0 + p;
    int sp = spos[i * NTOK + j];
    float sb = semb[sp * NH + l4];
    bias[((long)l4 << 20) + i * NTOK + j] = sb + (acc0[r] + acc1[r]) / pl;
  }
}

// ---------- 4. generic bf16 WMMA GEMM, 16x32 per wave (A reuse) ----------
// EPI: 0=Q(scale+head layout) 1=K(head layout) 2=V(transposed head layout)
//      3=f32 out + bias + residual      4=relu -> bf16 row-major
template <int EPI>
__global__ __launch_bounds__(256) void k_gemm(
    const bf16_t* __restrict__ A, int lda,
    const bf16_t* __restrict__ Bt, int ldb,
    const float* __restrict__ bias, const float* __restrict__ resid,
    float* __restrict__ outF, bf16_t* __restrict__ outB,
    int M, int Nout, int Kdim) {
  int lane = threadIdx.x & 31, wave = threadIdx.x >> 5;
  int m0 = blockIdx.x * 16;
  int n0 = blockIdx.y * 256 + wave * 32;       // 32 output cols per wave
  int hi = lane >> 4, l4 = lane & 15;
  const bf16_t* arow  = A + (long)(m0 + l4) * lda;
  const bf16_t* brow0 = Bt + (long)(n0 + l4) * ldb;
  const bf16_t* brow1 = Bt + (long)(n0 + 16 + l4) * ldb;
  v8f acc0 = {}, acc1 = {};
  for (int kk = 0; kk < Kdim; kk += 32) {
    __builtin_prefetch(arow + kk + 32, 0, 0);
    __builtin_prefetch(brow0 + kk + 32, 0, 0);
    __builtin_prefetch(brow1 + kk + 32, 0, 0);
    v16bf a  = load_frag_b(arow, kk, hi);      // shared A fragment
    v16bf b0 = load_frag_b(brow0, kk, hi);
    v16bf b1 = load_frag_b(brow1, kk, hi);
    acc0 = wmma_bf16(a, b0, acc0);
    acc1 = wmma_bf16(a, b1, acc1);
  }
#pragma unroll
  for (int r = 0; r < 8; ++r) {
    int row = m0 + r + (hi ? 8 : 0);
#pragma unroll
    for (int t = 0; t < 2; ++t) {
      int col = n0 + t * 16 + l4;
      float v = (t ? acc1[r] : acc0[r]) + bias[col];
      if constexpr (EPI == 0) {
        v *= 5.656854249492381f;  // sqrt(DH)
        int hh = col >> 5, d = col & 31;
        outB[((long)hh * M + row) * DH + d] = (bf16_t)v;
      } else if constexpr (EPI == 1) {
        int hh = col >> 5, d = col & 31;
        outB[((long)hh * M + row) * DH + d] = (bf16_t)v;
      } else if constexpr (EPI == 2) {
        int hh = col >> 5, d = col & 31;
        outB[((long)hh * DH + d) * M + row] = (bf16_t)v;  // V transposed [h][d][j]
      } else if constexpr (EPI == 3) {
        outF[(long)row * Nout + col] = v + resid[(long)row * Nout + col];
      } else if constexpr (EPI == 4) {
        outB[(long)row * Nout + col] = (bf16_t)(v > 0.f ? v : 0.f);
      }
    }
  }
}

// ---------- 5. flash attention per (head, 16-query tile) wave ----------
__global__ __launch_bounds__(256) void k_flash(
    const bf16_t* __restrict__ q, const bf16_t* __restrict__ k,
    const bf16_t* __restrict__ vT, const float* __restrict__ bias,
    bf16_t* __restrict__ obf) {
  __shared__ __align__(16) bf16_t pbuf[8][16 * 40];  // P transpose tile per wave (padded)
  int lane = threadIdx.x & 31, wave = threadIdx.x >> 5;
  int task = blockIdx.x * 8 + wave;  // 16 heads * 64 tiles
  int head = task >> 6, i0 = (task & 63) << 4;
  int hi = lane >> 4, l4 = lane & 15;
  const bf16_t* qh = q + ((long)head << 15);
  const bf16_t* kh = k + ((long)head << 15);
  const bf16_t* vh = vT + ((long)head << 15);
  const float*  bh = bias + ((long)head << 20);
  bf16_t* pb = pbuf[wave];

  v16bf qa = load_frag_b(qh + (long)(i0 + l4) * DH, 0, hi);
  float mrow[8], lrow[8];
  v8f o0 = {}, o1 = {};
#pragma unroll
  for (int r = 0; r < 8; ++r) { mrow[r] = -1e30f; lrow[r] = 0.f; }

  for (int jt = 0; jt < NTOK; jt += 32) {
    v16bf kb0 = load_frag_b(kh + (long)(jt + l4) * DH, 0, hi);
    v16bf kb1 = load_frag_b(kh + (long)(jt + 16 + l4) * DH, 0, hi);
    v8f s0 = {}, s1 = {};
    s0 = wmma_bf16(qa, kb0, s0);
    s1 = wmma_bf16(qa, kb1, s1);
#pragma unroll
    for (int r = 0; r < 8; ++r) {
      int row = i0 + r + (hi ? 8 : 0);
      const float* bp = bh + ((long)row << 10) + jt + l4;
      s0[r] += bp[0];
      s1[r] += bp[16];
    }
#pragma unroll
    for (int r = 0; r < 8; ++r) {
      float mx = fmaxf(s0[r], s1[r]);
#pragma unroll
      for (int d = 8; d >= 1; d >>= 1) mx = fmaxf(mx, __shfl_xor(mx, d, 16));
      float mn = fmaxf(mrow[r], mx);
      float scale = __expf(mrow[r] - mn);
      float p0 = __expf(s0[r] - mn);
      float p1 = __expf(s1[r] - mn);
      float sum = p0 + p1;
#pragma unroll
      for (int d = 8; d >= 1; d >>= 1) sum += __shfl_xor(sum, d, 16);
      lrow[r] = lrow[r] * scale + sum;
      mrow[r] = mn;
      o0[r] *= scale;
      o1[r] *= scale;
      int prw = r + (hi ? 8 : 0);
      pb[prw * 40 + l4] = (bf16_t)p0;        // D-layout -> LDS
      pb[prw * 40 + 16 + l4] = (bf16_t)p1;
    }
    asm volatile("s_wait_dscnt 0" ::: "memory");  // same-wave LDS RAW
    v16bf pa  = load_frag_b(pb + l4 * 40, 0, hi);               // A-layout from LDS
    v16bf vb0 = load_frag_b(vh + (long)l4 * NTOK, jt, hi);      // V^T rows d=0..15
    v16bf vb1 = load_frag_b(vh + (long)(16 + l4) * NTOK, jt, hi);
    o0 = wmma_bf16(pa, vb0, o0);
    o1 = wmma_bf16(pa, vb1, o1);
  }
#pragma unroll
  for (int r = 0; r < 8; ++r) {
    int row = i0 + r + (hi ? 8 : 0);
    float inv = 1.f / lrow[r];
    obf[(long)row * HID + head * DH + l4] = (bf16_t)(o0[r] * inv);
    obf[(long)row * HID + head * DH + 16 + l4] = (bf16_t)(o1[r] * inv);
  }
}

// ---------- 6. layernorm (row of 512) ----------
__global__ __launch_bounds__(256) void k_ln(const float* __restrict__ in,
                                            const float* __restrict__ g, const float* __restrict__ b,
                                            float* __restrict__ outF, bf16_t* __restrict__ outB) {
  __shared__ float red[256];
  int n = blockIdx.x, t = threadIdx.x;
  float v0 = in[n * HID + t], v1 = in[n * HID + 256 + t];
  red[t] = v0 + v1;
  __syncthreads();
  for (int s = 128; s > 0; s >>= 1) { if (t < s) red[t] += red[t + s]; __syncthreads(); }
  float mu = red[0] / (float)HID;
  __syncthreads();
  float d0 = v0 - mu, d1 = v1 - mu;
  red[t] = d0 * d0 + d1 * d1;
  __syncthreads();
  for (int s = 128; s > 0; s >>= 1) { if (t < s) red[t] += red[t + s]; __syncthreads(); }
  float rstd = rsqrtf(red[0] / (float)HID + 1e-5f);
  float y0 = d0 * rstd * g[t] + b[t];
  float y1 = d1 * rstd * g[256 + t] + b[256 + t];
  outF[n * HID + t] = y0;
  outF[n * HID + 256 + t] = y1;
  if (outB) {
    outB[n * HID + t] = (bf16_t)y0;
    outB[n * HID + 256 + t] = (bf16_t)y1;
  }
}

// ---------- launch ----------
extern "C" void kernel_launch(void* const* d_in, const int* in_sizes, int n_in,
                              void* d_out, int out_size, void* d_ws, size_t ws_size,
                              hipStream_t stream) {
  const float* x     = (const float*)d_in[0];
  const int*  spos   = (const int*)d_in[1];
  const int*  eidx   = (const int*)d_in[2];
  const int*  indeg  = (const int*)d_in[3];
  const int*  outdeg = (const int*)d_in[4];
  const float* semb  = (const float*)d_in[5];
  const float* eemb  = (const float*)d_in[6];
  const float* epw   = (const float*)d_in[7];
  const float* inde  = (const float*)d_in[8];
  const float* outde = (const float*)d_in[9];
  const float* Wq = (const float*)d_in[10]; const float* bq = (const float*)d_in[11];
  const float* Wk = (const float*)d_in[12]; const float* bk = (const float*)d_in[13];
  const float* Wv = (const float*)d_in[14]; const float* bv = (const float*)d_in[15];
  const float* Wo = (const float*)d_in[16]; const float* bo = (const float*)d_in[17];
  const float* W1 = (const float*)d_in[18]; const float* b1 = (const float*)d_in[19];
  const float* W2 = (const float*)d_in[20]; const float* b2 = (const float*)d_in[21];
  const float* g1 = (const float*)d_in[22]; const float* be1 = (const float*)d_in[23];
  const float* g2 = (const float*)d_in[24]; const float* be2 = (const float*)d_in[25];
  float* out = (float*)d_out;

  char* base = (char*)d_ws;
  size_t off = 0;
  auto carve = [&](size_t bytes) -> char* {
    char* p = base + off;
    off += (bytes + 255) & ~(size_t)255;
    return p;
  };
  float*  h     = (float*)carve((size_t)NTOK * HID * 4);
  bf16_t* hbf   = (bf16_t*)carve((size_t)NTOK * HID * 2);
  bf16_t* WqT   = (bf16_t*)carve((size_t)HID * HID * 2);
  bf16_t* WkT   = (bf16_t*)carve((size_t)HID * HID * 2);
  bf16_t* WvT   = (bf16_t*)carve((size_t)HID * HID * 2);
  bf16_t* WoT   = (bf16_t*)carve((size_t)HID * HID * 2);
  bf16_t* W1T   = (bf16_t*)carve((size_t)HID * 2 * HID * 2);
  bf16_t* W2T   = (bf16_t*)carve((size_t)HID * 2 * HID * 2);
  bf16_t* pwT   = (bf16_t*)carve((size_t)NH * 128 * 2);
  bf16_t* qb    = (bf16_t*)carve((size_t)NTOK * HID * 2);
  bf16_t* kb    = (bf16_t*)carve((size_t)NTOK * HID * 2);
  bf16_t* vTb   = (bf16_t*)carve((size_t)NTOK * HID * 2);
  float*  biasb = (float*)carve((size_t)NH * NTOK * NTOK * 4);  // 64 MB
  bf16_t* obf   = (bf16_t*)carve((size_t)NTOK * HID * 2);
  float*  res1  = (float*)carve((size_t)NTOK * HID * 4);
  float*  h1    = (float*)carve((size_t)NTOK * HID * 4);
  bf16_t* h1bf  = (bf16_t*)carve((size_t)NTOK * HID * 2);
  bf16_t* midbf = (bf16_t*)carve((size_t)NTOK * 2 * HID * 2);
  float*  res2  = (float*)carve((size_t)NTOK * HID * 4);

  // prep
  k_centrality<<<NTOK, 128, 0, stream>>>(x, indeg, outdeg, inde, outde, h, hbf);
  k_wt<<<(HID * HID + 255) / 256, 256, 0, stream>>>(Wq, WqT, HID, HID);
  k_wt<<<(HID * HID + 255) / 256, 256, 0, stream>>>(Wk, WkT, HID, HID);
  k_wt<<<(HID * HID + 255) / 256, 256, 0, stream>>>(Wv, WvT, HID, HID);
  k_wt<<<(HID * HID + 255) / 256, 256, 0, stream>>>(Wo, WoT, HID, HID);
  k_wt<<<(HID * 2 * HID + 255) / 256, 256, 0, stream>>>(W1, W1T, HID, 2 * HID);
  k_wt<<<(HID * 2 * HID + 255) / 256, 256, 0, stream>>>(W2, W2T, 2 * HID, HID);
  k_pw<<<(NH * 128 + 255) / 256, 256, 0, stream>>>(epw, pwT);

  // attention bias (WMMA einsum + spatial gather)
  k_bias<<<(NTOK * 64) / 8, 256, 0, stream>>>(spos, eidx, semb, eemb, pwT, biasb);

  // q/k/v projections (16x32 tiles per wave)
  k_gemm<0><<<dim3(64, 2), 256, 0, stream>>>(hbf, HID, WqT, HID, bq, nullptr, nullptr, qb, NTOK, HID, HID);
  k_gemm<1><<<dim3(64, 2), 256, 0, stream>>>(hbf, HID, WkT, HID, bk, nullptr, nullptr, kb, NTOK, HID, HID);
  k_gemm<2><<<dim3(64, 2), 256, 0, stream>>>(hbf, HID, WvT, HID, bv, nullptr, nullptr, vTb, NTOK, HID, HID);

  // flash attention
  k_flash<<<(NH * 64) / 8, 256, 0, stream>>>(qb, kb, vTb, biasb, obf);

  // output projection + residual, LN1
  k_gemm<3><<<dim3(64, 2), 256, 0, stream>>>(obf, HID, WoT, HID, bo, h, res1, nullptr, NTOK, HID, HID);
  k_ln<<<NTOK, 256, 0, stream>>>(res1, g1, be1, h1, h1bf);

  // FFN + residual, LN2 -> out
  k_gemm<4><<<dim3(64, 4), 256, 0, stream>>>(h1bf, HID, W1T, HID, b1, nullptr, nullptr, midbf, NTOK, 2 * HID, HID);
  k_gemm<3><<<dim3(64, 2), 256, 0, stream>>>(midbf, 2 * HID, W2T, 2 * HID, b2, h1, res2, nullptr, NTOK, HID, 2 * HID);
  k_ln<<<NTOK, 256, 0, stream>>>(res2, g2, be2, out, nullptr);
}
